// Transformer_39350490365974
// MI455X (gfx1250) — matrix operand order repaired and checked by reading, not compiled, not run on hardware
//
#include <hip/hip_runtime.h>
#include <hip/hip_bf16.h>

typedef __attribute__((ext_vector_type(16))) __bf16 v16bf;
typedef __attribute__((ext_vector_type(8)))  float  v8f;

namespace {
constexpr int kB   = 4;
constexpr int kS   = 1024;
constexpr int kD   = 512;
constexpr int kH   = 8;
constexpr int kDK  = 64;
constexpr int kDFF = 2048;
constexpr int kL   = 6;
constexpr int kV   = 16000;
constexpr int kM   = kB * kS;   // 4096 rows
}

// ---------------------------------------------------------------------------
// Embedding gather + positional encoding
// ---------------------------------------------------------------------------
__global__ __launch_bounds__(256) void embed_pe_kernel(const int* __restrict__ tok,
                                                       const float* __restrict__ emb,
                                                       float* __restrict__ out) {
  int rs = blockIdx.x;           // b*S + s
  int s  = rs & (kS - 1);
  int t  = tok[rs];
  for (int d = threadIdx.x; d < kD; d += blockDim.x) {
    float ex  = (float)(d & ~1) / (float)kD;
    float div = __powf(10000.0f, ex);
    float ang = (float)s / div;
    float pe  = (d & 1) ? __cosf(ang) : __sinf(ang);
    out[(size_t)rs * kD + d] = emb[(size_t)t * kD + d] + pe;
  }
}

// ---------------------------------------------------------------------------
// WMMA bf16 GEMM: C[M,N] = A[M,K] * W[K,N] + bias  (optional ReLU)
// macro tile 128x128, K-step 32; 256 threads = 8 waves, wave owns 16x128.
// Tiles staged fp32 in LDS via GLOBAL_LOAD_ASYNC_TO_LDS_B128, double-buffered;
// bf16 conversion happens at fragment build.
// ---------------------------------------------------------------------------
#define AS_STRIDE 36
#define BS_STRIDE 132

template <bool RELU>
__global__ __launch_bounds__(256) void gemm_kernel(const float* __restrict__ A,
                                                   const float* __restrict__ W,
                                                   const float* __restrict__ bias,
                                                   float* __restrict__ C,
                                                   int M, int N, int K) {
  __shared__ float As[2][128][AS_STRIDE];  // 128x32 fp32 tile (+pad)
  __shared__ float Bs[2][32][BS_STRIDE];   // 32x128 fp32 tile (+pad)

  const int tid  = threadIdx.x;
  const int lane = tid & 31;
  const int wave = tid >> 5;
  const int m0 = blockIdx.y << 7;
  const int n0 = blockIdx.x << 7;

  v8f acc[8];
#pragma unroll
  for (int i = 0; i < 8; ++i)
#pragma unroll
    for (int j = 0; j < 8; ++j) acc[i][j] = 0.0f;

  const int ar = tid >> 1, ac = (tid & 1) << 4;  // A: 16 floats / thread
  const int wr = tid >> 3, wc = (tid & 7) << 4;  // W: 16 floats / thread

  const unsigned asBase = (unsigned)(uintptr_t)(&As[0][0][0]);
  const unsigned bsBase = (unsigned)(uintptr_t)(&Bs[0][0][0]);
  const unsigned asBuf  = 128u * AS_STRIDE * 4u;
  const unsigned bsBuf  = 32u * BS_STRIDE * 4u;
  const unsigned aLds0  = asBase + (unsigned)(ar * AS_STRIDE + ac) * 4u;
  const unsigned bLds0  = bsBase + (unsigned)(wr * BS_STRIDE + wc) * 4u;

  // Issue 8x b128 async copies (64B of A + 64B of W per thread) for one K-step.
  auto issue_tile = [&](int k0, int buf) {
    unsigned ga = (unsigned)(((size_t)(m0 + ar) * K + k0 + ac) * 4u);
    unsigned la = aLds0 + (unsigned)buf * asBuf;
#pragma unroll
    for (int i = 0; i < 4; ++i) {
      asm volatile("global_load_async_to_lds_b128 %0, %1, %2"
                   :: "v"(la + 16u * i), "v"(ga + 16u * i), "s"(A)
                   : "memory");
    }
    unsigned gw = (unsigned)(((size_t)(k0 + wr) * N + n0 + wc) * 4u);
    unsigned lb = bLds0 + (unsigned)buf * bsBuf;
#pragma unroll
    for (int i = 0; i < 4; ++i) {
      asm volatile("global_load_async_to_lds_b128 %0, %1, %2"
                   :: "v"(lb + 16u * i), "v"(gw + 16u * i), "s"(W)
                   : "memory");
    }
  };

  const int arow = (wave << 4) + (lane & 15);
  const int akb  = (lane >> 4) << 4;
  const int bcol = lane & 15;
  const int bkb  = (lane >> 4) << 4;

  int buf = 0;
  issue_tile(0, 0);

  for (int k0 = 0; k0 < K; k0 += 32) {
    asm volatile("s_wait_asynccnt 0x0" ::: "memory");  // own tile landed
    __syncthreads();                                    // everyone's tile landed
    if (k0 + 32 < K) issue_tile(k0 + 32, buf ^ 1);      // prefetch next tile

    v16bf af;
#pragma unroll
    for (int e = 0; e < 16; ++e)
      af[e] = (__bf16)As[buf][arow][akb + e];

#pragma unroll
    for (int ns = 0; ns < 8; ++ns) {
      v16bf bf_;
#pragma unroll
      for (int e = 0; e < 16; ++e)
        bf_[e] = (__bf16)Bs[buf][bkb + e][(ns << 4) + bcol];
      acc[ns] = __builtin_amdgcn_wmma_f32_16x16x32_bf16(
          false, af, false, bf_, (short)0, acc[ns], false, false);
    }
    buf ^= 1;
  }

  const int mrow = m0 + (wave << 4) + ((lane >> 4) << 3);
  const int ncol = n0 + (lane & 15);
#pragma unroll
  for (int ns = 0; ns < 8; ++ns) {
    int n = ncol + (ns << 4);
    float bv = bias[n];
#pragma unroll
    for (int j = 0; j < 8; ++j) {
      float c = acc[ns][j] + bv;
      if (RELU) c = fmaxf(c, 0.0f);
      C[(size_t)(mrow + j) * N + n] = c;
    }
  }
}

// ---------------------------------------------------------------------------
// Fused flash attention (per (b,h)), heads packed in [B,S,D] at col h*DK.
// Block = 128 threads = 4 waves; each wave owns 16 query rows.
// Row-sum of P computed as P x ones via an extra WMMA (no shuffle tree).
// ---------------------------------------------------------------------------
__global__ __launch_bounds__(128) void attn_kernel(const float* __restrict__ Qt,
                                                   const float* __restrict__ Kt,
                                                   const float* __restrict__ Vt,
                                                   float* __restrict__ O,
                                                   int causal) {
  __shared__ float pbuf[4][16][36];
  __shared__ float vbuf[4][32][68];

  const int lane = threadIdx.x & 31;
  const int wave = threadIdx.x >> 5;
  const int bh = blockIdx.y;
  const int b = bh / kH, h = bh % kH;
  const int qbase = (blockIdx.x << 6) + (wave << 4);
  const size_t base = (size_t)b * kS * kD + (size_t)h * kDK;

  // Q fragments (scale 1/sqrt(DK)=0.125 folded in)
  v16bf aq[2];
  {
    const int qr = qbase + (lane & 15);
    const int kb = (lane >> 4) << 4;
    const float* qp = Qt + base + (size_t)qr * kD;
#pragma unroll
    for (int c = 0; c < 2; ++c)
#pragma unroll
      for (int e = 0; e < 16; ++e) {
        int kk = (c << 5) + kb + e;
        aq[c][e] = (__bf16)(qp[kk] * 0.125f);
      }
  }

  v16bf bones;  // all-ones B fragment for row-sum WMMA
#pragma unroll
  for (int e = 0; e < 16; ++e) bones[e] = (__bf16)1.0f;

  float mrow[8], lrow[8];
  v8f acc[4];
#pragma unroll
  for (int j = 0; j < 8; ++j) { mrow[j] = -1e30f; lrow[j] = 0.0f; }
#pragma unroll
  for (int d = 0; d < 4; ++d)
#pragma unroll
    for (int j = 0; j < 8; ++j) acc[d][j] = 0.0f;

  const int nblocks = causal ? (qbase >> 5) + 1 : (kS >> 5);

  for (int jb = 0; jb < nblocks; ++jb) {
    const int k0 = jb << 5;

    // ---- scores: two 16x16 tiles (32 keys), K-dim 64 = 2 chunks ----
    v8f s0, s1;
#pragma unroll
    for (int j = 0; j < 8; ++j) { s0[j] = 0.0f; s1[j] = 0.0f; }
#pragma unroll
    for (int g = 0; g < 2; ++g) {
      const int key = k0 + (g << 4) + (lane & 15);
      const float* kp = Kt + base + (size_t)key * kD;
      v8f sg;
#pragma unroll
      for (int j = 0; j < 8; ++j) sg[j] = 0.0f;
#pragma unroll
      for (int c = 0; c < 2; ++c) {
        v16bf bk;
        const int dbase = (c << 5) + ((lane >> 4) << 4);
        const float4* kp4 = (const float4*)(kp + dbase);
#pragma unroll
        for (int i = 0; i < 4; ++i) {
          float4 f = kp4[i];
          bk[i * 4 + 0] = (__bf16)f.x;
          bk[i * 4 + 1] = (__bf16)f.y;
          bk[i * 4 + 2] = (__bf16)f.z;
          bk[i * 4 + 3] = (__bf16)f.w;
        }
        sg = __builtin_amdgcn_wmma_f32_16x16x32_bf16(
            false, aq[c], false, bk, (short)0, sg, false, false);
      }
      if (g == 0) s0 = sg; else s1 = sg;
    }

    // ---- causal mask (only for tail blocks) ----
    if (causal && (k0 + 31 > qbase)) {
      const int half = (lane >> 4) << 3;
      const int kc = lane & 15;
#pragma unroll
      for (int j = 0; j < 8; ++j) {
        int qr = qbase + j + half;
        if (k0 + kc > qr)      s0[j] = -1e30f;
        if (k0 + 16 + kc > qr) s1[j] = -1e30f;
      }
    }

    // ---- online softmax: row max via wave32 shuffle tree ----
    float alpha[8];
#pragma unroll
    for (int j = 0; j < 8; ++j) {
      float m_ = fmaxf(s0[j], s1[j]);
      m_ = fmaxf(m_, __shfl_xor(m_, 1));
      m_ = fmaxf(m_, __shfl_xor(m_, 2));
      m_ = fmaxf(m_, __shfl_xor(m_, 4));
      m_ = fmaxf(m_, __shfl_xor(m_, 8));
      float mn = fmaxf(mrow[j], m_);
      alpha[j] = __expf(mrow[j] - mn);
      mrow[j] = mn;
    }
#pragma unroll
    for (int j = 0; j < 8; ++j) {
      s0[j] = __expf(s0[j] - mrow[j]);
      s1[j] = __expf(s1[j] - mrow[j]);
    }
#pragma unroll
    for (int d = 0; d < 4; ++d)
#pragma unroll
      for (int j = 0; j < 8; ++j) acc[d][j] *= alpha[j];

    // ---- P (C-layout) -> LDS -> A-layout bf16 fragment ----
    {
      const int m_ = (lane >> 4) << 3;
      const int n_ = lane & 15;
#pragma unroll
      for (int j = 0; j < 8; ++j) {
        pbuf[wave][m_ + j][n_] = s0[j];
        pbuf[wave][m_ + j][n_ + 16] = s1[j];
      }
    }
    // ---- stage V block (32 keys x 64 dims), contiguous row loads ----
    {
      const float4* vp4 = (const float4*)(Vt + base + (size_t)(k0 + lane) * kD);
#pragma unroll
      for (int i = 0; i < 16; ++i) {
        float4 f = vp4[i];
        vbuf[wave][lane][i * 4 + 0] = f.x;
        vbuf[wave][lane][i * 4 + 1] = f.y;
        vbuf[wave][lane][i * 4 + 2] = f.z;
        vbuf[wave][lane][i * 4 + 3] = f.w;
      }
    }

    v16bf ap;
    {
      const int pr = lane & 15;
      const int kbp = (lane >> 4) << 4;
#pragma unroll
      for (int e = 0; e < 16; ++e)
        ap[e] = (__bf16)pbuf[wave][pr][kbp + e];
    }

    // ---- row-sum of P via WMMA against all-ones B (all cols identical) ----
    {
      v8f rs;
#pragma unroll
      for (int j = 0; j < 8; ++j) rs[j] = 0.0f;
      rs = __builtin_amdgcn_wmma_f32_16x16x32_bf16(
          false, ap, false, bones, (short)0, rs, false, false);
#pragma unroll
      for (int j = 0; j < 8; ++j) lrow[j] = lrow[j] * alpha[j] + rs[j];
    }

#pragma unroll
    for (int d = 0; d < 4; ++d) {
      v16bf bv;
      const int dim = (d << 4) + (lane & 15);
      const int kb2 = (lane >> 4) << 4;
#pragma unroll
      for (int e = 0; e < 16; ++e)
        bv[e] = (__bf16)vbuf[wave][kb2 + e][dim];
      acc[d] = __builtin_amdgcn_wmma_f32_16x16x32_bf16(
          false, ap, false, bv, (short)0, acc[d], false, false);
    }
  }

  // ---- normalize + store ----
  const int half = (lane >> 4) << 3;
#pragma unroll
  for (int d = 0; d < 4; ++d) {
    const int dim = (d << 4) + (lane & 15);
#pragma unroll
    for (int j = 0; j < 8; ++j) {
      int qr = qbase + j + half;
      O[base + (size_t)qr * kD + dim] = acc[d][j] / lrow[j];
    }
  }
}

// ---------------------------------------------------------------------------
// x = LayerNorm(x + a) * g + b  — one wave per row of D=512
// ---------------------------------------------------------------------------
__global__ __launch_bounds__(256) void add_ln_kernel(float* __restrict__ x,
                                                     const float* __restrict__ a,
                                                     const float* __restrict__ g,
                                                     const float* __restrict__ bb) {
  const int lane = threadIdx.x & 31;
  const int row = (blockIdx.x << 3) + (threadIdx.x >> 5);
  float v[16];
  float sum = 0.0f;
#pragma unroll
  for (int i = 0; i < 16; ++i) {
    int d = (i << 5) + lane;
    v[i] = x[(size_t)row * kD + d] + a[(size_t)row * kD + d];
    sum += v[i];
  }
#pragma unroll
  for (int off = 16; off >= 1; off >>= 1) sum += __shfl_xor(sum, off);
  const float mu = sum * (1.0f / kD);
  float var = 0.0f;
#pragma unroll
  for (int i = 0; i < 16; ++i) { float t = v[i] - mu; var += t * t; }
#pragma unroll
  for (int off = 16; off >= 1; off >>= 1) var += __shfl_xor(var, off);
  const float rstd = rsqrtf(var * (1.0f / kD) + 1e-5f);
#pragma unroll
  for (int i = 0; i < 16; ++i) {
    int d = (i << 5) + lane;
    x[(size_t)row * kD + d] = g[d] * (v[i] - mu) * rstd + bb[d];
  }
}

// ---------------------------------------------------------------------------
// Host orchestration
// ---------------------------------------------------------------------------
extern "C" void kernel_launch(void* const* d_in, const int* in_sizes, int n_in,
                              void* d_out, int out_size, void* d_ws, size_t ws_size,
                              hipStream_t stream) {
  (void)in_sizes; (void)n_in; (void)out_size; (void)ws_size;
  const int*   src       = (const int*)d_in[0];
  const int*   tgt       = (const int*)d_in[1];
  // d_in[2] = tgt_mask (causal computed inline)
  const float* src_emb   = (const float*)d_in[3];
  const float* tgt_emb   = (const float*)d_in[4];
  const float* e_att_w   = (const float*)d_in[5];
  const float* e_att_b   = (const float*)d_in[6];
  const float* e_w1      = (const float*)d_in[7];
  const float* e_b1      = (const float*)d_in[8];
  const float* e_w2      = (const float*)d_in[9];
  const float* e_b2      = (const float*)d_in[10];
  const float* e_ln_g    = (const float*)d_in[11];
  const float* e_ln_b    = (const float*)d_in[12];
  const float* d_att_w   = (const float*)d_in[13];
  const float* d_att_b   = (const float*)d_in[14];
  const float* d_w1      = (const float*)d_in[15];
  const float* d_b1      = (const float*)d_in[16];
  const float* d_w2      = (const float*)d_in[17];
  const float* d_b2      = (const float*)d_in[18];
  const float* d_ln_g    = (const float*)d_in[19];
  const float* d_ln_b    = (const float*)d_in[20];
  const float* gen_w     = (const float*)d_in[21];
  const float* gen_b     = (const float*)d_in[22];

  const size_t MD = (size_t)kM * kD;
  float* ws = (float*)d_ws;
  float* X  = ws;            // encoder state / memory
  float* Y  = X + MD;        // decoder state
  float* Qb = Y + MD;
  float* Kb = Qb + MD;
  float* Vb = Kb + MD;
  float* Ab = Vb + MD;       // attention output
  float* Tb = Ab + MD;       // residual branch
  float* Hb = Tb + MD;       // FFN hidden [M, DFF]

  auto gemm = [&](const float* A, const float* W, const float* bias, float* C,
                  int M, int N, int K, bool relu) {
    dim3 grid(N / 128, M / 128);
    if (relu)
      gemm_kernel<true><<<grid, 256, 0, stream>>>(A, W, bias, C, M, N, K);
    else
      gemm_kernel<false><<<grid, 256, 0, stream>>>(A, W, bias, C, M, N, K);
  };
  auto attn = [&](const float* q, const float* k, const float* v, float* o, int causal) {
    dim3 grid(kS / 64, kB * kH);
    attn_kernel<<<grid, 128, 0, stream>>>(q, k, v, o, causal);
  };
  auto addln = [&](float* x, const float* a, const float* g, const float* b) {
    add_ln_kernel<<<kM / 8, 256, 0, stream>>>(x, a, g, b);
  };

  const size_t DD = (size_t)kD * kD;

  // -------- Encoder --------
  embed_pe_kernel<<<kM, 256, 0, stream>>>(src, src_emb, X);
  for (int i = 0; i < kL; ++i) {
    const float* w = e_att_w + (size_t)i * 4 * DD;
    const float* bw = e_att_b + (size_t)i * 4 * kD;
    gemm(X, w + 0 * DD, bw + 0 * kD, Qb, kM, kD, kD, false);
    gemm(X, w + 1 * DD, bw + 1 * kD, Kb, kM, kD, kD, false);
    gemm(X, w + 2 * DD, bw + 2 * kD, Vb, kM, kD, kD, false);
    attn(Qb, Kb, Vb, Ab, 0);
    gemm(Ab, w + 3 * DD, bw + 3 * kD, Tb, kM, kD, kD, false);
    addln(X, Tb, e_ln_g + (size_t)(i * 2 + 0) * kD, e_ln_b + (size_t)(i * 2 + 0) * kD);
    gemm(X, e_w1 + (size_t)i * kD * kDFF, e_b1 + (size_t)i * kDFF, Hb, kM, kDFF, kD, true);
    gemm(Hb, e_w2 + (size_t)i * kDFF * kD, e_b2 + (size_t)i * kD, Tb, kM, kD, kDFF, false);
    addln(X, Tb, e_ln_g + (size_t)(i * 2 + 1) * kD, e_ln_b + (size_t)(i * 2 + 1) * kD);
  }

  // -------- Decoder --------
  embed_pe_kernel<<<kM, 256, 0, stream>>>(tgt, tgt_emb, Y);
  for (int i = 0; i < kL; ++i) {
    const float* w = d_att_w + (size_t)i * 8 * DD;
    const float* bw = d_att_b + (size_t)i * 8 * kD;
    // masked self-attention
    gemm(Y, w + 0 * DD, bw + 0 * kD, Qb, kM, kD, kD, false);
    gemm(Y, w + 1 * DD, bw + 1 * kD, Kb, kM, kD, kD, false);
    gemm(Y, w + 2 * DD, bw + 2 * kD, Vb, kM, kD, kD, false);
    attn(Qb, Kb, Vb, Ab, 1);
    gemm(Ab, w + 3 * DD, bw + 3 * kD, Tb, kM, kD, kD, false);
    addln(Y, Tb, d_ln_g + (size_t)(i * 3 + 0) * kD, d_ln_b + (size_t)(i * 3 + 0) * kD);
    // cross-attention (K,V from encoder memory X)
    gemm(Y, w + 4 * DD, bw + 4 * kD, Qb, kM, kD, kD, false);
    gemm(X, w + 5 * DD, bw + 5 * kD, Kb, kM, kD, kD, false);
    gemm(X, w + 6 * DD, bw + 6 * kD, Vb, kM, kD, kD, false);
    attn(Qb, Kb, Vb, Ab, 0);
    gemm(Ab, w + 7 * DD, bw + 7 * kD, Tb, kM, kD, kD, false);
    addln(Y, Tb, d_ln_g + (size_t)(i * 3 + 1) * kD, d_ln_b + (size_t)(i * 3 + 1) * kD);
    // FFN
    gemm(Y, d_w1 + (size_t)i * kD * kDFF, d_b1 + (size_t)i * kDFF, Hb, kM, kDFF, kD, true);
    gemm(Hb, d_w2 + (size_t)i * kDFF * kD, d_b2 + (size_t)i * kD, Tb, kM, kD, kDFF, false);
    addln(Y, Tb, d_ln_g + (size_t)(i * 3 + 2) * kD, d_ln_b + (size_t)(i * 3 + 2) * kD);
  }

  // -------- Generator --------
  gemm(Y, gen_w, gen_b, (float*)d_out, kM, kV, kD, false);
}
